// MoE_10222022165065
// MI455X (gfx1250) — compile-verified, matrix operand order checked by reading, not run on hardware
//
#include <hip/hip_runtime.h>
#include <hip/hip_bf16.h>

// MoE forward for MI455X (gfx1250): bf16 WMMA grouped GEMMs + shared expert.
// Pipeline: convert weights/x to bf16 -> router (top-2) -> histogram ->
// stable bucket ranks with 16-row padded segments -> gather/scale ->
// grouped up-GEMM (relu^2, bf16) -> grouped down-GEMM (bf16-rounded f32) ->
// shared up/down GEMMs -> scatter-sum combine.
// GEMM: 8 waves/block, each wave owns a 16x64 tile = 4 v_wmma_f32_16x16x32_bf16
// accumulators sharing one A fragment per k-step (raises FLOP/byte and WMMA
// density per fetch; bf16 weights are L2-resident so direct b128 loads win).

typedef __attribute__((ext_vector_type(16))) __bf16 v16bf;
typedef __attribute__((ext_vector_type(8)))  float  v8f;

#define BS_   1
#define SLEN_ 2048
#define T_    (BS_ * SLEN_)
#define DIM_  1024
#define HID_  1408
#define NEXP_ 8
#define TOPK_ 2
#define NSLOT_ (T_ * TOPK_)
// max padded rows: 4096 + 8*15 = 4216, rounded to 32 -> 4224 (264 m-tiles)
#define MCAP_ 4224

struct alignas(16) Q16 { unsigned long long lo, hi; };
union V16U { v16bf v; Q16 q2[2]; };

// ---------------------------------------------------------------- convert
__global__ void f32_to_bf16_kernel(const float* __restrict__ src,
                                   __hip_bfloat16* __restrict__ dst,
                                   long long n) {
    long long i = (long long)blockIdx.x * blockDim.x + threadIdx.x;
    long long stride = (long long)gridDim.x * blockDim.x;
    for (; i < n; i += stride) dst[i] = __float2bfloat16(src[i]);
}

// ---------------------------------------------------------------- router
__global__ void router_kernel(const float* __restrict__ x,
                              const float* __restrict__ gate_w,
                              const float* __restrict__ bias,
                              int* __restrict__ sel,
                              float* __restrict__ top_scores) {
    __shared__ float red[256];
    __shared__ float sc[NEXP_];
    const int t = blockIdx.x, tid = threadIdx.x;
    float acc[NEXP_];
#pragma unroll
    for (int e = 0; e < NEXP_; ++e) acc[e] = 0.f;
    for (int k = tid; k < DIM_; k += 256) {
        float xv = x[(long long)t * DIM_ + k];
#pragma unroll
        for (int e = 0; e < NEXP_; ++e) acc[e] += xv * gate_w[e * DIM_ + k];
    }
    for (int e = 0; e < NEXP_; ++e) {
        red[tid] = acc[e];
        __syncthreads();
        for (int s = 128; s > 0; s >>= 1) {
            if (tid < s) red[tid] += red[tid + s];
            __syncthreads();
        }
        if (tid == 0) sc[e] = 1.f / (1.f + __expf(-red[0]));
        __syncthreads();
    }
    if (tid == 0) {
        float b[NEXP_];
#pragma unroll
        for (int e = 0; e < NEXP_; ++e) b[e] = sc[e] + bias[e];
        int i1 = 0; float m1 = b[0];
#pragma unroll
        for (int e = 1; e < NEXP_; ++e) if (b[e] > m1) { m1 = b[e]; i1 = e; }
        int i2 = -1; float m2 = -3.0e38f;
#pragma unroll
        for (int e = 0; e < NEXP_; ++e)
            if (e != i1 && b[e] > m2) { m2 = b[e]; i2 = e; }
        sel[2 * t]     = i1;
        sel[2 * t + 1] = i2;
        top_scores[2 * t]     = sc[i1];
        top_scores[2 * t + 1] = sc[i2];
    }
}

// ---------------------------------------------------------------- dispatch
__global__ void zero_counts_kernel(int* counts) {
    if (threadIdx.x < NEXP_) counts[threadIdx.x] = 0;
}

__global__ void count_kernel(const int* __restrict__ sel, int* __restrict__ counts) {
    int i = blockIdx.x * blockDim.x + threadIdx.x;
    if (i < NSLOT_) atomicAdd(&counts[sel[i]], 1);
}

// 1 block, 32 threads. Thread 0 builds padded offsets; threads 0..7 each scan
// all slots in order to assign stable ranks for their expert (argsort-stable).
__global__ void rank_kernel(const int* __restrict__ sel,
                            const float* __restrict__ top_scores,
                            const int* __restrict__ counts,
                            int* __restrict__ poff,
                            int* __restrict__ src_token,
                            float* __restrict__ score_sorted,
                            int* __restrict__ dst_row) {
    __shared__ int sp[NEXP_ + 1];
    const int tid = threadIdx.x;
    if (tid == 0) {
        int acc = 0;
        for (int e = 0; e < NEXP_; ++e) { sp[e] = acc; acc += (counts[e] + 15) & ~15; }
        sp[NEXP_] = acc;
        for (int e = 0; e <= NEXP_; ++e) poff[e] = sp[e];
    }
    __syncthreads();
    if (tid < NEXP_) {
        const int e = tid;
        int r = sp[e];
        for (int i = 0; i < NSLOT_; ++i) {
            if (sel[i] == e) {
                src_token[r]    = i >> 1;          // token id
                score_sorted[r] = top_scores[i];
                dst_row[i]      = r;
                ++r;
            }
        }
        for (int end = sp[e + 1]; r < end; ++r) src_token[r] = -1;  // pad rows
    }
}

__global__ void gather_scale_kernel(const float* __restrict__ x,
                                    const int* __restrict__ src_token,
                                    const float* __restrict__ score_sorted,
                                    const int* __restrict__ poff,
                                    __hip_bfloat16* __restrict__ routed_in) {
    const int row = blockIdx.x;
    const int p8  = poff[NEXP_];
    const int s   = (row < p8) ? src_token[row] : -1;
    const float sc = (s >= 0) ? score_sorted[row] : 0.f;
    const float* xr = x + (long long)(s >= 0 ? s : 0) * DIM_;
    for (int c = threadIdx.x; c < DIM_; c += blockDim.x) {
        float v = (s >= 0) ? xr[c] * sc : 0.f;
        routed_in[(long long)row * DIM_ + c] = __float2bfloat16(v);
    }
}

// ---------------------------------------------------------------- WMMA GEMM
// C[M,N] = A[M,K] * W[N,K]^T, bf16 inputs, f32 accumulation via
// v_wmma_f32_16x16x32_bf16. Block = 256 threads = 8 waves; each wave owns a
// 16x64 output tile (four 16x16 WMMA accumulators sharing one A fragment).
// Block footprint: 32 rows x 256 cols. If poff != nullptr the GEMM is grouped:
// each 16-row M-tile maps to one expert (segments padded to 16 rows).
// act: 0 = raw f32 out, 1 = relu(bf16(x))^2 -> bf16 out, 2 = bf16-rounded f32.
__device__ __forceinline__ void load_frag(V16U& u, const __hip_bfloat16* base, int ka) {
    u.q2[0] = *(const Q16*)(base + ka);
    u.q2[1] = *(const Q16*)(base + ka + 16);
}

__global__ void wmma_gemm_kernel(const __hip_bfloat16* __restrict__ A,
                                 const __hip_bfloat16* __restrict__ W,
                                 float* __restrict__ Cf,
                                 __hip_bfloat16* __restrict__ Cb,
                                 const int* __restrict__ poff,
                                 int M, int N, int K, long long wstride, int act) {
    const int tid    = threadIdx.x;
    const int wave   = tid >> 5;
    const int lane   = tid & 31;
    const int lane15 = lane & 15;
    const int hi     = (lane >> 4) & 1;   // lane half selects K sub-block + row half

    const int m0 = (blockIdx.x * 2 + (wave >> 2)) * 16;
    const int n0 = blockIdx.y * 256 + (wave & 3) * 64;

    int e = 0;
    if (poff) {
        const int p8 = poff[NEXP_];
        if (m0 >= p8) return;
#pragma unroll
        for (int g = 1; g < NEXP_; ++g) if (m0 >= poff[g]) e = g;
    } else {
        if (m0 >= M) return;
    }
    if (n0 + 64 > N) return;   // wave-uniform; N is a multiple of 64

    const __hip_bfloat16* Wb   = W + (long long)e * wstride;
    const __hip_bfloat16* arow = A  + (long long)(m0 + lane15) * K;
    const __hip_bfloat16* w0   = Wb + (long long)(n0 + lane15) * K;
    const __hip_bfloat16* w1   = w0 + (long long)16 * K;
    const __hip_bfloat16* w2   = w0 + (long long)32 * K;
    const __hip_bfloat16* w3   = w0 + (long long)48 * K;

    v8f c0 = {}, c1 = {}, c2 = {}, c3 = {};
    const int kofs = hi * 8;  // ISA 16-bit A/B layout: lane half owns K 0-7/16-23 vs 8-15/24-31
    for (int k0 = 0; k0 < K; k0 += 32) {
        V16U a, b0, b1, b2, b3;
        const int ka = k0 + kofs;
        load_frag(a,  arow, ka);
        load_frag(b0, w0,   ka);
        load_frag(b1, w1,   ka);
        load_frag(b2, w2,   ka);
        load_frag(b3, w3,   ka);
        c0 = __builtin_amdgcn_wmma_f32_16x16x32_bf16(false, a.v, false, b0.v,
                                                     (short)0, c0, false, false);
        c1 = __builtin_amdgcn_wmma_f32_16x16x32_bf16(false, a.v, false, b1.v,
                                                     (short)0, c1, false, false);
        c2 = __builtin_amdgcn_wmma_f32_16x16x32_bf16(false, a.v, false, b2.v,
                                                     (short)0, c2, false, false);
        c3 = __builtin_amdgcn_wmma_f32_16x16x32_bf16(false, a.v, false, b3.v,
                                                     (short)0, c3, false, false);
    }

    // 16x16 f32 C/D layout: VGPR r, lanes 0-15 -> M = r, lanes 16-31 -> M = r+8.
    const int colb = n0 + lane15;
    float* accs[1];  // silence unused warnings pattern not needed; direct writes below
    (void)accs;
#pragma unroll
    for (int r = 0; r < 8; ++r) {
        const long long row = m0 + r + hi * 8;
        float v[4] = { c0[r], c1[r], c2[r], c3[r] };
#pragma unroll
        for (int j = 0; j < 4; ++j) {
            const long long idx = row * N + colb + 16 * j;
            float vv = v[j];
            if (act == 1) {
                vv = __bfloat162float(__float2bfloat16(vv));
                vv = vv > 0.f ? vv * vv : 0.f;
                Cb[idx] = __float2bfloat16(vv);
            } else if (act == 2) {
                Cf[idx] = __bfloat162float(__float2bfloat16(vv));
            } else {
                Cf[idx] = vv;
            }
        }
    }
}

// ---------------------------------------------------------------- combine
__global__ void combine_kernel(const float* __restrict__ shared_out,
                               const float* __restrict__ routed_out,
                               const int* __restrict__ dst_row,
                               float* __restrict__ out) {
    const int t = blockIdx.x;
    const long long r0 = dst_row[2 * t];
    const long long r1 = dst_row[2 * t + 1];
    for (int c = threadIdx.x; c < DIM_; c += blockDim.x) {
        out[(long long)t * DIM_ + c] =
            shared_out[(long long)t * DIM_ + c] +
            routed_out[r0 * DIM_ + c] + routed_out[r1 * DIM_ + c];
    }
}

// ---------------------------------------------------------------- launch
extern "C" void kernel_launch(void* const* d_in, const int* in_sizes, int n_in,
                              void* d_out, int out_size, void* d_ws, size_t ws_size,
                              hipStream_t stream) {
    (void)in_sizes; (void)n_in; (void)out_size; (void)ws_size;
    const float* x        = (const float*)d_in[0];  // (T, DIM)
    const float* gate_w   = (const float*)d_in[1];  // (E, DIM)
    const float* bias     = (const float*)d_in[2];  // (E,)
    const float* w_up     = (const float*)d_in[3];  // (E, HID, DIM)
    const float* w_down   = (const float*)d_in[4];  // (E, DIM, HID)
    const float* sh_up    = (const float*)d_in[5];  // (HID, DIM)
    const float* sh_down  = (const float*)d_in[6];  // (DIM, HID)
    float* out = (float*)d_out;                     // (T, DIM) f32

    char* w = (char*)d_ws;
    auto alloc = [&](size_t bytes) -> char* {
        char* p = w;
        w += (bytes + 255) & ~(size_t)255;
        return p;
    };
    const long long WUP_N = (long long)NEXP_ * HID_ * DIM_;
    const long long SH_N  = (long long)HID_ * DIM_;

    __hip_bfloat16* w_up_b   = (__hip_bfloat16*)alloc(WUP_N * 2);
    __hip_bfloat16* w_down_b = (__hip_bfloat16*)alloc(WUP_N * 2);
    __hip_bfloat16* sh_up_b  = (__hip_bfloat16*)alloc(SH_N * 2);
    __hip_bfloat16* sh_dn_b  = (__hip_bfloat16*)alloc(SH_N * 2);
    __hip_bfloat16* x_b      = (__hip_bfloat16*)alloc((long long)T_ * DIM_ * 2);
    __hip_bfloat16* routed_in= (__hip_bfloat16*)alloc((long long)MCAP_ * DIM_ * 2);
    __hip_bfloat16* h_buf    = (__hip_bfloat16*)alloc((long long)MCAP_ * HID_ * 2);
    float*          routed_o = (float*)alloc((long long)MCAP_ * DIM_ * 4);
    __hip_bfloat16* hs_buf   = (__hip_bfloat16*)alloc((long long)T_ * HID_ * 2);
    float*          shared_o = (float*)alloc((long long)T_ * DIM_ * 4);
    int*   sel          = (int*)alloc(NSLOT_ * 4);
    float* top_scores   = (float*)alloc(NSLOT_ * 4);
    int*   counts       = (int*)alloc(NEXP_ * 4);
    int*   poff         = (int*)alloc((NEXP_ + 1) * 4);
    int*   src_token    = (int*)alloc(MCAP_ * 4);
    float* score_sorted = (float*)alloc(MCAP_ * 4);
    int*   dst_row      = (int*)alloc(NSLOT_ * 4);

    // 1. precision conversion (one pass; bf16 weights ~46MB -> resident in 192MB L2)
    f32_to_bf16_kernel<<<2048, 256, 0, stream>>>(w_up,    w_up_b,   WUP_N);
    f32_to_bf16_kernel<<<2048, 256, 0, stream>>>(w_down,  w_down_b, WUP_N);
    f32_to_bf16_kernel<<<1024, 256, 0, stream>>>(sh_up,   sh_up_b,  SH_N);
    f32_to_bf16_kernel<<<1024, 256, 0, stream>>>(sh_down, sh_dn_b,  SH_N);
    f32_to_bf16_kernel<<<1024, 256, 0, stream>>>(x,       x_b, (long long)T_ * DIM_);

    // 2. routing
    router_kernel<<<T_, 256, 0, stream>>>(x, gate_w, bias, sel, top_scores);
    zero_counts_kernel<<<1, 32, 0, stream>>>(counts);
    count_kernel<<<(NSLOT_ + 255) / 256, 256, 0, stream>>>(sel, counts);
    rank_kernel<<<1, 32, 0, stream>>>(sel, top_scores, counts, poff,
                                      src_token, score_sorted, dst_row);
    gather_scale_kernel<<<MCAP_, 256, 0, stream>>>(x, src_token, score_sorted,
                                                   poff, routed_in);

    // 3. grouped expert MLP (bf16 WMMA, f32 accum)
    wmma_gemm_kernel<<<dim3(MCAP_ / 32, (HID_ + 255) / 256), 256, 0, stream>>>(
        routed_in, w_up_b, nullptr, h_buf, poff,
        MCAP_, HID_, DIM_, (long long)HID_ * DIM_, /*act=*/1);
    wmma_gemm_kernel<<<dim3(MCAP_ / 32, DIM_ / 256), 256, 0, stream>>>(
        h_buf, w_down_b, routed_o, nullptr, poff,
        MCAP_, DIM_, HID_, (long long)DIM_ * HID_, /*act=*/2);

    // 4. shared expert
    wmma_gemm_kernel<<<dim3(T_ / 32, (HID_ + 255) / 256), 256, 0, stream>>>(
        x_b, sh_up_b, nullptr, hs_buf, nullptr,
        T_, HID_, DIM_, 0, /*act=*/1);
    wmma_gemm_kernel<<<dim3(T_ / 32, DIM_ / 256), 256, 0, stream>>>(
        hs_buf, sh_dn_b, shared_o, nullptr, nullptr,
        T_, DIM_, HID_, 0, /*act=*/0);

    // 5. combine
    combine_kernel<<<T_, 256, 0, stream>>>(shared_o, routed_o, dst_row, out);
}
